// BLiqNet_15101105013368
// MI455X (gfx1250) — matrix operand order repaired and checked by reading, not compiled, not run on hardware
//
#include <hip/hip_runtime.h>
#include <math.h>

// ---------------------------------------------------------------------------
// Fused Liquid-ODE network for MI455X (gfx1250, wave32, WMMA bf16).
//   u     = x @ Wx^T + bx
//   drive = u @ U^T + b          (f32, parked in LDS)
//   40 RK4 steps of dh/dt = (-h + tanh(h @ W^T + drive)) / tau
//   out   = h @ Wf^T + bf
// One workgroup per 32 batch rows. h and the RK4 accumulator live in WMMA
// C-fragment registers; the bf16 A-stage (32x512, 32 KB) and drive (64 KB)
// live in LDS. Weights are pre-packed bf16 B-fragments, L2-resident.
// ---------------------------------------------------------------------------

typedef __attribute__((ext_vector_type(16))) __bf16 bf16x16;
typedef __attribute__((ext_vector_type(8)))  __bf16 bf16x8;
typedef __attribute__((ext_vector_type(8)))  float  f32x8;

#define HID     512
#define INDIM   256
#define OUTDIM  128
#define MT      32          // batch rows per workgroup
#define THREADS 512         // 16 wave32s
#define NSTEPS  40

__device__ __forceinline__ float fast_tanh(float v) {
#if __has_builtin(__builtin_amdgcn_tanhf)
  return __builtin_amdgcn_tanhf(v);   // V_TANH_F32 (gfx1250 trans op)
#else
  return tanhf(v);
#endif
}

// ---------------------------------------------------------------------------
// Pack a row-major [N][K] f32 weight into WMMA B-fragment order (bf16).
// Lane l of tile (kt,nt) holds 16 bf16: K = kt*32 + (l/16)*8 + (j%8) + (j/8)*16,
// N = nt*16 + (l%16)   (ISA 16-bit B 32x16 striping).
// ---------------------------------------------------------------------------
__global__ void pack_b_kernel(const float* __restrict__ src,
                              __bf16* __restrict__ dst, int K, int N) {
  int idx = blockIdx.x * 256 + threadIdx.x;
  if (idx >= K * N) return;
  int j    = idx & 15;
  int l    = (idx >> 4) & 31;
  int tile = idx >> 9;
  int Nt   = N >> 4;
  int kt   = tile / Nt;
  int nt   = tile - kt * Nt;
  int k    = kt * 32 + ((l >> 4) << 3) + (j & 7) + ((j >> 3) << 4);
  int n    = nt * 16 + (l & 15);
  dst[idx] = (__bf16)src[(size_t)n * K + k];
}

// ---------------------------------------------------------------------------
// One wave computes a 16(M) x 64(N) f32 tile: acc += stage(MxK) @ Bpacked.
// ---------------------------------------------------------------------------
__device__ __forceinline__ void wave_gemm(const __bf16* __restrict__ Bp,
                                          const __bf16* __restrict__ st,
                                          int m0, int n0t, int K, int Ntiles,
                                          f32x8 acc[4]) {
  const int lane  = threadIdx.x & 31;
  const int abase = (m0 + (lane & 15)) * HID + ((lane >> 4) << 3);
  const int kts   = K >> 5;
  for (int kt = 0; kt < kts; ++kt) {
    const __bf16* ap = st + abase + (kt << 5);
    bf16x8 alo = *(const bf16x8*)(ap);
    bf16x8 ahi = *(const bf16x8*)(ap + 16);
    bf16x16 a  = __builtin_shufflevector(alo, ahi,
                   0, 1, 2, 3, 4, 5, 6, 7, 8, 9, 10, 11, 12, 13, 14, 15);
    const __bf16* bb = Bp + (((size_t)(kt * Ntiles + n0t) * 32 + lane) << 4);
    if (kt + 1 < kts)  // keep next K-step's B tiles hot (global_prefetch_b8)
      __builtin_prefetch(
          Bp + (((size_t)((kt + 1) * Ntiles + n0t) * 32 + lane) << 4), 0, 3);
#pragma unroll
    for (int s = 0; s < 4; ++s) {
      bf16x16 b = *(const bf16x16*)(bb + ((size_t)s << 9));  // +s*32*16 elems
      acc[s] = __builtin_amdgcn_wmma_f32_16x16x32_bf16(
                 false, a, false, b, (short)0, acc[s], false, false);
    }
  }
}

__global__ __launch_bounds__(THREADS)
void liquid_ode_fused(const float* __restrict__ x,
                      const float* __restrict__ bx,
                      const float* __restrict__ bvec,
                      const float* __restrict__ tau,
                      const float* __restrict__ bfh,
                      const __bf16* __restrict__ WxP,
                      const __bf16* __restrict__ UP,
                      const __bf16* __restrict__ WP,
                      const __bf16* __restrict__ WfP,
                      float* __restrict__ out) {
  __shared__ __bf16 stage[MT * HID];   // 32 KB bf16 A-matrix stage
  __shared__ float  driveS[MT * HID];  // 64 KB constant input drive

  const int  tid  = threadIdx.x;
  const int  lane = tid & 31;
  const int  w    = tid >> 5;
  const int  m0   = (w >> 3) << 4;  // 0 or 16 (local row tile)
  const int  n0   = (w & 7) << 6;   // 0..448  (hidden col tile)
  const int  n0t  = (w & 7) << 2;   // col tile index in 16-col units
  const long gm0  = (long)blockIdx.x * MT;

  // Per-lane fragment coordinates (C/D layout).
  const int mb = m0 + ((lane >> 4) << 3);  // row base, rows mb..mb+7
  const int cb = n0 + (lane & 15);         // col base, cols cb + s*16

  // Stage x tile as bf16 (K = 0..255; row stride stays HID).
  for (int e = tid; e < MT * INDIM; e += THREADS) {
    int m = e >> 8, k = e & (INDIM - 1);
    stage[m * HID + k] = (__bf16)x[(gm0 + m) * INDIM + k];
  }

  float itau[4];
#pragma unroll
  for (int s = 0; s < 4; ++s) itau[s] = 1.0f / tau[cb + s * 16];
  __syncthreads();

  const f32x8 z8 = {0.f, 0.f, 0.f, 0.f, 0.f, 0.f, 0.f, 0.f};
  f32x8 acc[4];

  // ---- u = x @ Wx^T + bx  (h0 = u) ----
#pragma unroll
  for (int s = 0; s < 4; ++s) acc[s] = z8;
  wave_gemm(WxP, stage, m0, n0t, INDIM, HID / 16, acc);
  float h[4][8];
#pragma unroll
  for (int s = 0; s < 4; ++s) {
    float bias = bx[cb + s * 16];
#pragma unroll
    for (int i = 0; i < 8; ++i) h[s][i] = acc[s][i] + bias;
  }
  __syncthreads();  // all waves done reading x-stage
#pragma unroll
  for (int s = 0; s < 4; ++s)
#pragma unroll
    for (int i = 0; i < 8; ++i)
      stage[(mb + i) * HID + cb + s * 16] = (__bf16)h[s][i];  // bf16(u)
  __syncthreads();

  // ---- drive = u @ U^T + b  -> LDS ----
#pragma unroll
  for (int s = 0; s < 4; ++s) acc[s] = z8;
  wave_gemm(UP, stage, m0, n0t, HID, HID / 16, acc);
#pragma unroll
  for (int s = 0; s < 4; ++s) {
    float bias = bvec[cb + s * 16];
#pragma unroll
    for (int i = 0; i < 8; ++i)
      driveS[(mb + i) * HID + cb + s * 16] = acc[s][i] + bias;
  }
  __syncthreads();
  // Invariant entering each eval: stage == bf16(h_in), barrier passed.

  const float dt    = 1.0f / NSTEPS;
  const float wc[4] = {dt / 6.f, dt / 3.f, dt / 3.f, dt / 6.f};
  const float sc[3] = {0.5f * dt, 0.5f * dt, dt};

#pragma unroll 1
  for (int step = 0; step < NSTEPS; ++step) {
    float hn[4][8];
#pragma unroll
    for (int s = 0; s < 4; ++s)
#pragma unroll
      for (int i = 0; i < 8; ++i) hn[s][i] = h[s][i];

#pragma unroll
    for (int e = 0; e < 4; ++e) {
#pragma unroll
      for (int s = 0; s < 4; ++s) acc[s] = z8;
      wave_gemm(WP, stage, m0, n0t, HID, HID / 16, acc);  // h_in @ W^T

      // k = (tanh(z + drive) - h_in) / tau; h_in re-read from bf16 stage
      // (exactly what the GEMM consumed). kv lives only across one barrier.
      float kv[4][8];
#pragma unroll
      for (int s = 0; s < 4; ++s)
#pragma unroll
        for (int i = 0; i < 8; ++i) {
          int   idx = (mb + i) * HID + cb + s * 16;
          float hin = (float)stage[idx];
          float g   = fast_tanh(acc[s][i] + driveS[idx]);
          kv[s][i]  = (g - hin) * itau[s];
          hn[s][i] += wc[e] * kv[s][i];
        }
      if (e < 3) {
        __syncthreads();  // everyone finished reading old stage
#pragma unroll
        for (int s = 0; s < 4; ++s)
#pragma unroll
          for (int i = 0; i < 8; ++i)
            stage[(mb + i) * HID + cb + s * 16] =
                (__bf16)(h[s][i] + sc[e] * kv[s][i]);
        __syncthreads();
      }
    }
    // commit h and restore invariant stage == bf16(h)
#pragma unroll
    for (int s = 0; s < 4; ++s)
#pragma unroll
      for (int i = 0; i < 8; ++i) h[s][i] = hn[s][i];
    __syncthreads();
#pragma unroll
    for (int s = 0; s < 4; ++s)
#pragma unroll
      for (int i = 0; i < 8; ++i)
        stage[(mb + i) * HID + cb + s * 16] = (__bf16)h[s][i];
    __syncthreads();
  }

  // ---- out = h @ Wf^T + bf  (only N-tiles 0..127 exist) ----
  if ((w & 7) < 2) {
#pragma unroll
    for (int s = 0; s < 4; ++s) acc[s] = z8;
    wave_gemm(WfP, stage, m0, (w & 7) << 2, HID, OUTDIM / 16, acc);
    const int ocb = ((w & 7) << 6) + (lane & 15);
#pragma unroll
    for (int s = 0; s < 4; ++s) {
      int   c    = ocb + s * 16;
      float bias = bfh[c];
#pragma unroll
      for (int i = 0; i < 8; ++i)
        out[(gm0 + mb + i) * OUTDIM + c] = acc[s][i] + bias;
    }
  }
}

// ---------------------------------------------------------------------------
extern "C" void kernel_launch(void* const* d_in, const int* in_sizes, int n_in,
                              void* d_out, int out_size, void* d_ws,
                              size_t ws_size, hipStream_t stream) {
  const float* x   = (const float*)d_in[0];
  const float* Wx  = (const float*)d_in[1];
  const float* bx  = (const float*)d_in[2];
  const float* W   = (const float*)d_in[3];
  const float* U   = (const float*)d_in[4];
  const float* b   = (const float*)d_in[5];
  const float* tau = (const float*)d_in[6];
  const float* Wf  = (const float*)d_in[7];
  const float* bf_ = (const float*)d_in[8];
  float*       out = (float*)d_out;

  // Workspace layout: packed bf16 weights in WMMA B-fragment order.
  char*   ws  = (char*)d_ws;
  __bf16* WxP = (__bf16*)(ws);                                   // 256 KB
  __bf16* UP  = (__bf16*)(ws + (size_t)512 * 256 * 2);           // 512 KB
  __bf16* WP  = (__bf16*)(ws + (size_t)512 * 256 * 2 + (size_t)512 * 512 * 2);
  __bf16* WfP = (__bf16*)(ws + (size_t)512 * 256 * 2 + 2 * (size_t)512 * 512 * 2);

  pack_b_kernel<<<(512 * 256 + 255) / 256, 256, 0, stream>>>(Wx, WxP, 256, 512);
  pack_b_kernel<<<(512 * 512 + 255) / 256, 256, 0, stream>>>(U,  UP,  512, 512);
  pack_b_kernel<<<(512 * 512 + 255) / 256, 256, 0, stream>>>(W,  WP,  512, 512);
  pack_b_kernel<<<(512 * 128 + 255) / 256, 256, 0, stream>>>(Wf, WfP, 512, 128);

  const int batch = in_sizes[0] / INDIM;  // 32768
  liquid_ode_fused<<<batch / MT, THREADS, 0, stream>>>(
      x, bx, b, tau, bf_, WxP, UP, WP, WfP, out);
}
